// WaveNet_24352464568643
// MI455X (gfx1250) — matrix operand order repaired and checked by reading, not compiled, not run on hardware
//
#include <hip/hip_runtime.h>
#include <math.h>

#define TT 32000
#define NB 2
#define NT 64
#define BPB (TT / NT)          // 500 time tiles per batch
#define GRIDG (NB * BPB)       // 1000 blocks for GEMM kernels
#define RES 128
#define GATE 256
#define SKIPC 128
#define CINP 80
#define CPAD 96
#define OUTC 256
#define NLAY 20
#define SQH 0.70710678118654752440f

// LDS strides (elements). All chosen so col*stride*2 bytes is a multiple of 32
// -> every v16bf LDS load is 32B-aligned; all vector copies stay 16B-aligned.
#define S128 144   // 128-channel tiles (144*2 = 288 = 9*32)
#define S96  112   // 96-channel conditioning tile (224 = 7*32)
#define S256 272   // 256-channel tile for first conv (544 = 17*32)
#define HSTR 264   // fp32 h staging stride (264%64==8 -> spreads banks)

typedef __attribute__((ext_vector_type(16))) __bf16 v16bf;
typedef __attribute__((ext_vector_type(8)))  float  v8f;

__device__ __forceinline__ v8f wmma_bf16(v16bf a, v16bf b, v8f c) {
  // D = A(16x32 bf16) x B(32x16 bf16) + C(16x16 f32)
  return __builtin_amdgcn_wmma_f32_16x16x32_bf16(false, a, false, b, (short)0, c,
                                                 false, false);
}
__device__ __forceinline__ v8f zero8() {
  v8f z;
#pragma unroll
  for (int i = 0; i < 8; ++i) z[i] = 0.f;
  return z;
}
__device__ __forceinline__ unsigned short f2bf(float f) {
  unsigned u = __float_as_uint(f);
  u += 0x7FFFu + ((u >> 16) & 1u);   // round-to-nearest-even
  return (unsigned short)(u >> 16);
}
__device__ __forceinline__ unsigned pk2(float lo, float hi) {
  return (unsigned)f2bf(lo) | ((unsigned)f2bf(hi) << 16);
}
__device__ __forceinline__ float bf2f(unsigned short h) {
  return __uint_as_float(((unsigned)h) << 16);
}
__device__ __forceinline__ float frcp(float x) {
#if __has_builtin(__builtin_amdgcn_rcpf)
  return __builtin_amdgcn_rcpf(x);
#else
  return 1.f / x;
#endif
}
// fast, NaN-safe gate nonlinearities via v_exp_f32 / v_rcp_f32 TRANS ops
__device__ __forceinline__ float ftanh(float a) {
  float e = __expf(2.f * a);
  return 1.f - 2.f * frcp(e + 1.f);   // inf -> 1, 0 -> -1
}
__device__ __forceinline__ float fsigm(float g) {
  return frcp(1.f + __expf(-g));
}
// A fragment: pre-swizzled in global memory, 512 bf16 per 16x32 tile,
// lane-major so each lane reads 16 contiguous bf16 (32B).
__device__ __forceinline__ v16bf ldA(const unsigned short* f, int tile, int lane) {
  return *(const v16bf*)(f + (size_t)tile * 512 + lane * 16);
}
// B fragment from LDS tile stored [time][channel]: lane = column (time),
// 16 consecutive K (channels) per lane, split across lane halves.
__device__ __forceinline__ v16bf ldB(const unsigned short* lds, int stride, int col,
                                     int kb) {
  return *(const v16bf*)(lds + col * stride + kb);
}

// ---------------- prep kernels (coalesced LDS transposes) ----------------
__global__ void k_prep_x(const float* __restrict__ x, unsigned short* __restrict__ xo) {
  // x: [B][256][T] fp32 -> xo: [B][T][256] bf16 ; one 64t x 256ch tile per block
  __shared__ unsigned short lt[256 * 68];
  int tid = threadIdx.x;
  int blk = blockIdx.x, b = blk / BPB, t0 = (blk % BPB) * NT;
  size_t base = (size_t)b * TT + t0;
  for (int e = tid; e < 256 * NT; e += 256) {
    int ch = e >> 6, t = e & 63;  // coalesced along t
    lt[ch * 68 + t] = f2bf(x[((size_t)b * OUTC + ch) * TT + t0 + t]);
  }
  __syncthreads();
  for (int e = tid; e < NT * 256; e += 256) {
    int t = e >> 8, ch = e & 255;  // coalesced along ch
    xo[(base + t) * 256 + ch] = lt[ch * 68 + t];
  }
}
__global__ void k_prep_c(const float* __restrict__ c, unsigned short* __restrict__ co) {
  // c: [B][80][T] fp32 -> co: [B][T][96] bf16 (zero padded channels 80..95)
  __shared__ unsigned short lt[80 * 68];
  int tid = threadIdx.x;
  int blk = blockIdx.x, b = blk / BPB, t0 = (blk % BPB) * NT;
  size_t base = (size_t)b * TT + t0;
  for (int e = tid; e < 80 * NT; e += 256) {
    int ch = e >> 6, t = e & 63;
    lt[ch * 68 + t] = f2bf(c[((size_t)b * CINP + ch) * TT + t0 + t]);
  }
  __syncthreads();
  for (int e = tid; e < NT * CPAD; e += 256) {
    int t = e / CPAD, ch = e - t * CPAD;
    co[(base + t) * CPAD + ch] = (ch < CINP) ? lt[ch * 68 + t] : (unsigned short)0;
  }
}
// Swizzle fp32 weights [M][K(taps interleaved)] into WMMA A-fragment order.
// tile = mt*(taps*Kchunks) + tap*Kchunks + kc ; dst[tile][lane][e].
__global__ void k_swizzle(const float* __restrict__ src, unsigned short* __restrict__ dst,
                          int Kchunks, int taps, int Kin, int srcMstride, int srcKstride) {
  int tile = blockIdx.x;
  int kc = tile % Kchunks;
  int tmp = tile / Kchunks;
  int tap = tmp % taps;
  int mt = tmp / taps;
  for (int e2 = threadIdx.x; e2 < 512; e2 += 256) {
    int lane = e2 >> 4, e = e2 & 15;
    int m = mt * 16 + (lane & 15);
    // contiguous K order matching ldB: lanes 0-15 K=0..15, lanes 16-31 K=16..31
    int koff = (lane >> 4) * 16 + e;
    int k = kc * 32 + koff;
    float v = (k < Kin) ? src[(size_t)m * srcMstride + (size_t)k * srcKstride + tap] : 0.f;
    dst[(size_t)tile * 512 + e2] = f2bf(v);
  }
}

// ---------------- first 1x1 conv: x(256ch) -> res(128ch) bf16 ----------------
__global__ void __launch_bounds__(256) k_first(const unsigned short* __restrict__ xin,
                                               const unsigned short* __restrict__ wf,
                                               const float* __restrict__ bias,
                                               unsigned short* __restrict__ xout) {
  extern __shared__ unsigned char smem[];
  unsigned short* xs = (unsigned short*)smem;  // 64*272 bf16 = 34816B
  int tid = threadIdx.x, w = tid >> 5, lane = tid & 31;
  int blk = blockIdx.x, b = blk / BPB, t0 = (blk % BPB) * NT;
  size_t base = (size_t)b * TT + t0;
  for (int e = tid; e < NT * 32; e += 256) {  // 8xbf16 vector copies
    int t = e >> 5, c4 = e & 31;
    *(uint4*)(xs + t * S256 + c4 * 8) = *(const uint4*)(xin + (base + t) * 256 + c4 * 8);
  }
  __syncthreads();
  int mt = w;  // 8 mtiles, one per wave
  v8f acc[4];
#pragma unroll
  for (int nt = 0; nt < 4; ++nt) acc[nt] = zero8();
  int kbh = (lane >> 4) * 16, nl = lane & 15;
  for (int kc = 0; kc < 8; ++kc) {
    v16bf a = ldA(wf, mt * 8 + kc, lane);
    int kb = kc * 32 + kbh;
    v16bf bfr[4];
#pragma unroll
    for (int nt = 0; nt < 4; ++nt) bfr[nt] = ldB(xs, S256, nt * 16 + nl, kb);
#pragma unroll
    for (int nt = 0; nt < 4; ++nt) acc[nt] = wmma_bf16(a, bfr[nt], acc[nt]);
  }
  __syncthreads();
  unsigned short* ox = xs;  // reuse: 64*144 bf16
  int mloc = 8 * (lane >> 4);
  int mb = mt * 16 + mloc;
  float4 bb0 = *(const float4*)(bias + mb);
  float4 bb1 = *(const float4*)(bias + mb + 4);
#pragma unroll
  for (int nt = 0; nt < 4; ++nt) {
    int t = nt * 16 + nl;
    uint4 pv;
    pv.x = pk2(acc[nt][0] + bb0.x, acc[nt][1] + bb0.y);
    pv.y = pk2(acc[nt][2] + bb0.z, acc[nt][3] + bb0.w);
    pv.z = pk2(acc[nt][4] + bb1.x, acc[nt][5] + bb1.y);
    pv.w = pk2(acc[nt][6] + bb1.z, acc[nt][7] + bb1.w);
    *(uint4*)(ox + t * S128 + mb) = pv;
  }
  __syncthreads();
  for (int e = tid; e < NT * 16; e += 256) {
    int t = e >> 4, c4 = e & 15;
    *(uint4*)(xout + (base + t) * RES + c4 * 8) = *(const uint4*)(ox + t * S128 + c4 * 8);
  }
}

// ---------------- fused residual layer ----------------
__global__ void __launch_bounds__(256) k_layer(
    const unsigned short* __restrict__ xcur, unsigned short* __restrict__ xnext,
    const unsigned short* __restrict__ c96, float* __restrict__ skip,
    const unsigned short* __restrict__ wconv, const unsigned short* __restrict__ wcond,
    const unsigned short* __restrict__ wskip, const unsigned short* __restrict__ wout,
    const float* __restrict__ conv_b, const float* __restrict__ cond_b,
    const float* __restrict__ skip_b, const float* __restrict__ out_b, int dil,
    float skip_scale, int first_layer) {
  extern __shared__ unsigned char smem[];
  unsigned short* xs = (unsigned short*)smem;                        // 3*64*144 = 55296B
  unsigned short* cs = (unsigned short*)(smem + 55296);              // 64*112*2 = 14336B
  float* hs = (float*)(smem + 55296 + 14336);                        // 64*264*4 = 67584B
  unsigned short* zs = (unsigned short*)(smem + 55296 + 14336 + 67584);  // 18432B
  int tid = threadIdx.x, w = tid >> 5, lane = tid & 31;
  int blk = blockIdx.x, b = blk / BPB, t0 = (blk % BPB) * NT;
  size_t base = (size_t)b * TT + t0;

  // stage 3 dilated x taps (tap k reads x[t-(2-k)*dil]) as [t][ch] bf16, 16B chunks
  for (int e = tid; e < 3 * NT * 16; e += 256) {
    int tap = e >> 10;          // NT*16 = 1024
    int rem = e & 1023;
    int i = rem >> 4, c4 = rem & 15;
    int ts = t0 + i - (2 - tap) * dil;
    uint4 v = {0u, 0u, 0u, 0u};
    if (ts >= 0) v = *(const uint4*)(xcur + ((size_t)b * TT + ts) * RES + c4 * 8);
    *(uint4*)(xs + tap * (NT * S128) + i * S128 + c4 * 8) = v;
  }
  for (int e = tid; e < NT * 12; e += 256) {  // 96 ch = 12 x 8bf16
    int i = e / 12, c4 = e - i * 12;
    *(uint4*)(cs + i * S96 + c4 * 8) = *(const uint4*)(c96 + (base + i) * CPAD + c4 * 8);
  }
  __syncthreads();

  // GEMM1: h(256 x 64) = conv taps + conditioning ; wave handles mtiles {w, w+8}
  v8f a0c[4], a1c[4];
#pragma unroll
  for (int nt = 0; nt < 4; ++nt) { a0c[nt] = zero8(); a1c[nt] = zero8(); }
  int kbh = (lane >> 4) * 16, nl = lane & 15;
  for (int tap = 0; tap < 3; ++tap) {
    const unsigned short* xsp = xs + tap * (NT * S128);
#pragma unroll
    for (int kc = 0; kc < 4; ++kc) {
      v16bf a0 = ldA(wconv, w * 12 + tap * 4 + kc, lane);
      v16bf a1 = ldA(wconv, (w + 8) * 12 + tap * 4 + kc, lane);
      int kb = kc * 32 + kbh;
      v16bf bfr[4];
#pragma unroll
      for (int nt = 0; nt < 4; ++nt) bfr[nt] = ldB(xsp, S128, nt * 16 + nl, kb);
#pragma unroll
      for (int nt = 0; nt < 4; ++nt) a0c[nt] = wmma_bf16(a0, bfr[nt], a0c[nt]);
#pragma unroll
      for (int nt = 0; nt < 4; ++nt) a1c[nt] = wmma_bf16(a1, bfr[nt], a1c[nt]);
    }
  }
#pragma unroll
  for (int kc = 0; kc < 3; ++kc) {  // conditioning, K padded to 96
    v16bf a0 = ldA(wcond, w * 3 + kc, lane);
    v16bf a1 = ldA(wcond, (w + 8) * 3 + kc, lane);
    int kb = kc * 32 + kbh;
    v16bf bfr[4];
#pragma unroll
    for (int nt = 0; nt < 4; ++nt) bfr[nt] = ldB(cs, S96, nt * 16 + nl, kb);
#pragma unroll
    for (int nt = 0; nt < 4; ++nt) a0c[nt] = wmma_bf16(a0, bfr[nt], a0c[nt]);
#pragma unroll
    for (int nt = 0; nt < 4; ++nt) a1c[nt] = wmma_bf16(a1, bfr[nt], a1c[nt]);
  }
  // spill h to LDS [t][256] (D layout: vgpr v -> M = v + 8*(lane>=16), N = lane&15)
  int mloc = 8 * (lane >> 4);
#pragma unroll
  for (int nt = 0; nt < 4; ++nt) {
    int row = nt * 16 + nl;
    float* h0 = hs + row * HSTR + w * 16 + mloc;
    float* h1 = hs + row * HSTR + (w + 8) * 16 + mloc;
    *(float4*)h0 = make_float4(a0c[nt][0], a0c[nt][1], a0c[nt][2], a0c[nt][3]);
    *(float4*)(h0 + 4) = make_float4(a0c[nt][4], a0c[nt][5], a0c[nt][6], a0c[nt][7]);
    *(float4*)h1 = make_float4(a1c[nt][0], a1c[nt][1], a1c[nt][2], a1c[nt][3]);
    *(float4*)(h1 + 4) = make_float4(a1c[nt][4], a1c[nt][5], a1c[nt][6], a1c[nt][7]);
  }
  __syncthreads();
  // gating: z = tanh(a+bias) * sigmoid(g+bias), 4 channels per iteration
  for (int e = tid; e < NT * 32; e += 256) {
    int t = e >> 5, ch = (e & 31) * 4;
    float4 av = *(const float4*)(hs + t * HSTR + ch);
    float4 gv = *(const float4*)(hs + t * HSTR + ch + 128);
    float4 ab = *(const float4*)(conv_b + ch);
    float4 acb = *(const float4*)(cond_b + ch);
    float4 gb = *(const float4*)(conv_b + 128 + ch);
    float4 gcb = *(const float4*)(cond_b + 128 + ch);
    float z0 = ftanh(av.x + ab.x + acb.x) * fsigm(gv.x + gb.x + gcb.x);
    float z1 = ftanh(av.y + ab.y + acb.y) * fsigm(gv.y + gb.y + gcb.y);
    float z2 = ftanh(av.z + ab.z + acb.z) * fsigm(gv.z + gb.z + gcb.z);
    float z3 = ftanh(av.w + ab.w + acb.w) * fsigm(gv.w + gb.w + gcb.w);
    *(uint2*)(zs + t * S128 + ch) = make_uint2(pk2(z0, z1), pk2(z2, z3));
  }
  __syncthreads();

  // GEMM2: waves 0-3 compute out-conv (o), waves 4-7 skip-conv (s); 128x64 each
  int sel = w >> 2;
  int mtb = w & 3;
  const unsigned short* wf = sel ? wskip : wout;
  const float* bias2 = sel ? skip_b : out_b;
  v8f b0[4], b1[4];
#pragma unroll
  for (int nt = 0; nt < 4; ++nt) { b0[nt] = zero8(); b1[nt] = zero8(); }
#pragma unroll
  for (int kc = 0; kc < 4; ++kc) {
    v16bf a0 = ldA(wf, mtb * 4 + kc, lane);
    v16bf a1 = ldA(wf, (mtb + 4) * 4 + kc, lane);
    int kb = kc * 32 + kbh;
    v16bf bfr[4];
#pragma unroll
    for (int nt = 0; nt < 4; ++nt) bfr[nt] = ldB(zs, S128, nt * 16 + nl, kb);
#pragma unroll
    for (int nt = 0; nt < 4; ++nt) b0[nt] = wmma_bf16(a0, bfr[nt], b0[nt]);
#pragma unroll
    for (int nt = 0; nt < 4; ++nt) b1[nt] = wmma_bf16(a1, bfr[nt], b1[nt]);
  }
  unsigned short* ox = (unsigned short*)smem;   // reuse xs region: 64*144 bf16
  float* ss = (float*)(smem + 18432);           // 64*136 fp32 (fits in xs region)
  int m0b = mtb * 16 + mloc;
  int m1b = (mtb + 4) * 16 + mloc;
  float4 bz0a = *(const float4*)(bias2 + m0b);
  float4 bz0b = *(const float4*)(bias2 + m0b + 4);
  float4 bz1a = *(const float4*)(bias2 + m1b);
  float4 bz1b = *(const float4*)(bias2 + m1b + 4);
#pragma unroll
  for (int nt = 0; nt < 4; ++nt) {
    int t = nt * 16 + nl;
    float r0[8], r1[8];
    r0[0] = b0[nt][0] + bz0a.x; r0[1] = b0[nt][1] + bz0a.y;
    r0[2] = b0[nt][2] + bz0a.z; r0[3] = b0[nt][3] + bz0a.w;
    r0[4] = b0[nt][4] + bz0b.x; r0[5] = b0[nt][5] + bz0b.y;
    r0[6] = b0[nt][6] + bz0b.z; r0[7] = b0[nt][7] + bz0b.w;
    r1[0] = b1[nt][0] + bz1a.x; r1[1] = b1[nt][1] + bz1a.y;
    r1[2] = b1[nt][2] + bz1a.z; r1[3] = b1[nt][3] + bz1a.w;
    r1[4] = b1[nt][4] + bz1b.x; r1[5] = b1[nt][5] + bz1b.y;
    r1[6] = b1[nt][6] + bz1b.z; r1[7] = b1[nt][7] + bz1b.w;
    if (sel == 0) {  // out conv -> residual update (8xbf16 vector loads/stores)
      uint4 rv0 = *(const uint4*)(xcur + (base + t) * RES + m0b);
      uint4 rv1 = *(const uint4*)(xcur + (base + t) * RES + m1b);
      const unsigned short* rp0 = (const unsigned short*)&rv0;
      const unsigned short* rp1 = (const unsigned short*)&rv1;
      uint4 p0, p1;
      p0.x = pk2((r0[0] + bf2f(rp0[0])) * SQH, (r0[1] + bf2f(rp0[1])) * SQH);
      p0.y = pk2((r0[2] + bf2f(rp0[2])) * SQH, (r0[3] + bf2f(rp0[3])) * SQH);
      p0.z = pk2((r0[4] + bf2f(rp0[4])) * SQH, (r0[5] + bf2f(rp0[5])) * SQH);
      p0.w = pk2((r0[6] + bf2f(rp0[6])) * SQH, (r0[7] + bf2f(rp0[7])) * SQH);
      p1.x = pk2((r1[0] + bf2f(rp1[0])) * SQH, (r1[1] + bf2f(rp1[1])) * SQH);
      p1.y = pk2((r1[2] + bf2f(rp1[2])) * SQH, (r1[3] + bf2f(rp1[3])) * SQH);
      p1.z = pk2((r1[4] + bf2f(rp1[4])) * SQH, (r1[5] + bf2f(rp1[5])) * SQH);
      p1.w = pk2((r1[6] + bf2f(rp1[6])) * SQH, (r1[7] + bf2f(rp1[7])) * SQH);
      *(uint4*)(ox + t * S128 + m0b) = p0;
      *(uint4*)(ox + t * S128 + m1b) = p1;
    } else {  // skip conv, pre-scaled accumulation (float4 stores)
      float* s0 = ss + t * 136 + m0b;
      float* s1 = ss + t * 136 + m1b;
      *(float4*)s0 = make_float4(r0[0] * skip_scale, r0[1] * skip_scale,
                                 r0[2] * skip_scale, r0[3] * skip_scale);
      *(float4*)(s0 + 4) = make_float4(r0[4] * skip_scale, r0[5] * skip_scale,
                                       r0[6] * skip_scale, r0[7] * skip_scale);
      *(float4*)s1 = make_float4(r1[0] * skip_scale, r1[1] * skip_scale,
                                 r1[2] * skip_scale, r1[3] * skip_scale);
      *(float4*)(s1 + 4) = make_float4(r1[4] * skip_scale, r1[5] * skip_scale,
                                       r1[6] * skip_scale, r1[7] * skip_scale);
    }
  }
  __syncthreads();
  for (int e = tid; e < NT * 16; e += 256) {  // residual out, 8xbf16 chunks
    int t = e >> 4, c4 = e & 15;
    *(uint4*)(xnext + (base + t) * RES + c4 * 8) = *(const uint4*)(ox + t * S128 + c4 * 8);
  }
  for (int e = tid; e < NT * 32; e += 256) {  // skip accumulate, float4 chunks
    int t = e >> 5, c4 = e & 31;
    float4 sv = *(const float4*)(ss + t * 136 + c4 * 4);
    float* dst = skip + (base + t) * RES + c4 * 4;
    if (!first_layer) {
      float4 p = *(const float4*)dst;
      sv.x += p.x; sv.y += p.y; sv.z += p.z; sv.w += p.w;
    }
    *(float4*)dst = sv;
  }
}

// ------- fused head: relu -> 1x1(128) -> relu -> 1x1(256) -> fp32 [B][256][T] -------
__global__ void __launch_bounds__(256) k_head(const float* __restrict__ skip,
                                              const unsigned short* __restrict__ w1,
                                              const float* __restrict__ b1,
                                              const unsigned short* __restrict__ w2,
                                              const float* __restrict__ b2,
                                              float* __restrict__ out) {
  extern __shared__ unsigned char smem[];
  unsigned short* zs = (unsigned short*)smem;             // 64*144*2 = 18432B
  unsigned short* mid = (unsigned short*)(smem + 18432);  // 64*144*2 = 18432B
  float* os = (float*)(smem + 36864);                     // 256*68*4 = 69632B
  int tid = threadIdx.x, w = tid >> 5, lane = tid & 31;
  int blk = blockIdx.x, b = blk / BPB, t0 = (blk % BPB) * NT;
  size_t base = (size_t)b * TT + t0;
  for (int e = tid; e < NT * 32; e += 256) {  // relu + bf16 pack, 4 ch per iter
    int t = e >> 5, ch = (e & 31) * 4;
    float4 sv = *(const float4*)(skip + (base + t) * SKIPC + ch);
    *(uint2*)(zs + t * S128 + ch) =
        make_uint2(pk2(fmaxf(sv.x, 0.f), fmaxf(sv.y, 0.f)),
                   pk2(fmaxf(sv.z, 0.f), fmaxf(sv.w, 0.f)));
  }
  __syncthreads();
  int kbh = (lane >> 4) * 16, nl = lane & 15, mloc = 8 * (lane >> 4);
  // GEMM A: y1 = relu(W1 @ relu(skip) + b1), M=128, wave mt = w
  {
    v8f acc[4];
#pragma unroll
    for (int nt = 0; nt < 4; ++nt) acc[nt] = zero8();
#pragma unroll
    for (int kc = 0; kc < 4; ++kc) {
      v16bf a = ldA(w1, w * 4 + kc, lane);
      int kb = kc * 32 + kbh;
      v16bf bfr[4];
#pragma unroll
      for (int nt = 0; nt < 4; ++nt) bfr[nt] = ldB(zs, S128, nt * 16 + nl, kb);
#pragma unroll
      for (int nt = 0; nt < 4; ++nt) acc[nt] = wmma_bf16(a, bfr[nt], acc[nt]);
    }
    int mb = w * 16 + mloc;
    float4 bb0 = *(const float4*)(b1 + mb);
    float4 bb1 = *(const float4*)(b1 + mb + 4);
#pragma unroll
    for (int nt = 0; nt < 4; ++nt) {
      int t = nt * 16 + nl;
      uint4 pv;
      pv.x = pk2(fmaxf(acc[nt][0] + bb0.x, 0.f), fmaxf(acc[nt][1] + bb0.y, 0.f));
      pv.y = pk2(fmaxf(acc[nt][2] + bb0.z, 0.f), fmaxf(acc[nt][3] + bb0.w, 0.f));
      pv.z = pk2(fmaxf(acc[nt][4] + bb1.x, 0.f), fmaxf(acc[nt][5] + bb1.y, 0.f));
      pv.w = pk2(fmaxf(acc[nt][6] + bb1.z, 0.f), fmaxf(acc[nt][7] + bb1.w, 0.f));
      *(uint4*)(mid + t * S128 + mb) = pv;
    }
  }
  __syncthreads();
  // GEMM B: out = W2 @ y1 + b2, M=256, wave mtiles {w, w+8}
  {
    v8f c0[4], c1[4];
#pragma unroll
    for (int nt = 0; nt < 4; ++nt) { c0[nt] = zero8(); c1[nt] = zero8(); }
#pragma unroll
    for (int kc = 0; kc < 4; ++kc) {
      v16bf a0 = ldA(w2, w * 4 + kc, lane);
      v16bf a1 = ldA(w2, (w + 8) * 4 + kc, lane);
      int kb = kc * 32 + kbh;
      v16bf bfr[4];
#pragma unroll
      for (int nt = 0; nt < 4; ++nt) bfr[nt] = ldB(mid, S128, nt * 16 + nl, kb);
#pragma unroll
      for (int nt = 0; nt < 4; ++nt) c0[nt] = wmma_bf16(a0, bfr[nt], c0[nt]);
#pragma unroll
      for (int nt = 0; nt < 4; ++nt) c1[nt] = wmma_bf16(a1, bfr[nt], c1[nt]);
    }
    int m0b = w * 16 + mloc;
    int m1b = (w + 8) * 16 + mloc;
    float4 ba0 = *(const float4*)(b2 + m0b);
    float4 ba1 = *(const float4*)(b2 + m0b + 4);
    float4 bb0 = *(const float4*)(b2 + m1b);
    float4 bb1 = *(const float4*)(b2 + m1b + 4);
#pragma unroll
    for (int nt = 0; nt < 4; ++nt) {
      int t = nt * 16 + nl;
      os[(m0b + 0) * 68 + t] = c0[nt][0] + ba0.x;
      os[(m0b + 1) * 68 + t] = c0[nt][1] + ba0.y;
      os[(m0b + 2) * 68 + t] = c0[nt][2] + ba0.z;
      os[(m0b + 3) * 68 + t] = c0[nt][3] + ba0.w;
      os[(m0b + 4) * 68 + t] = c0[nt][4] + ba1.x;
      os[(m0b + 5) * 68 + t] = c0[nt][5] + ba1.y;
      os[(m0b + 6) * 68 + t] = c0[nt][6] + ba1.z;
      os[(m0b + 7) * 68 + t] = c0[nt][7] + ba1.w;
      os[(m1b + 0) * 68 + t] = c1[nt][0] + bb0.x;
      os[(m1b + 1) * 68 + t] = c1[nt][1] + bb0.y;
      os[(m1b + 2) * 68 + t] = c1[nt][2] + bb0.z;
      os[(m1b + 3) * 68 + t] = c1[nt][3] + bb0.w;
      os[(m1b + 4) * 68 + t] = c1[nt][4] + bb1.x;
      os[(m1b + 5) * 68 + t] = c1[nt][5] + bb1.y;
      os[(m1b + 6) * 68 + t] = c1[nt][6] + bb1.z;
      os[(m1b + 7) * 68 + t] = c1[nt][7] + bb1.w;
    }
  }
  __syncthreads();
  for (int e = tid; e < OUTC * 16; e += 256) {  // float4 coalesced output
    int ch = e >> 4, t4 = (e & 15) * 4;
    *(float4*)(out + (size_t)b * OUTC * TT + (size_t)ch * TT + t0 + t4) =
        *(const float4*)(os + ch * 68 + t4);
  }
}

// ---------------- host ----------------
extern "C" void kernel_launch(void* const* d_in, const int* in_sizes, int n_in,
                              void* d_out, int out_size, void* d_ws, size_t ws_size,
                              hipStream_t stream) {
  (void)in_sizes; (void)n_in; (void)out_size; (void)ws_size;
  // allow large dynamic LDS (harmless if not required)
  (void)hipFuncSetAttribute((const void*)k_layer,
                            hipFuncAttributeMaxDynamicSharedMemorySize, 155648);
  (void)hipFuncSetAttribute((const void*)k_head,
                            hipFuncAttributeMaxDynamicSharedMemorySize, 106496);

  const float* x_in = (const float*)d_in[0];
  const float* c_in = (const float*)d_in[1];
  const float* first_w = (const float*)d_in[2];
  const float* first_b = (const float*)d_in[3];
  const float* last1_w = (const float*)d_in[4];
  const float* last1_b = (const float*)d_in[5];
  const float* last2_w = (const float*)d_in[6];
  const float* last2_b = (const float*)d_in[7];

  unsigned char* ws = (unsigned char*)d_ws;
  size_t o = 0;
  auto take = [&](size_t bytes) {
    size_t r = o;
    o += (bytes + 255) & ~(size_t)255;
    return r;
  };
  const size_t SZ_X = (size_t)NB * TT * RES * 2;
  size_t oXA = take(SZ_X);
  size_t oXB = take(SZ_X);
  size_t oXIN = take((size_t)NB * TT * OUTC * 2);
  size_t oC = take((size_t)NB * TT * CPAD * 2);
  size_t oSK = take((size_t)NB * TT * SKIPC * 4);
  size_t oFF = take(64 * 1024);
  size_t oL1 = take(32 * 1024);
  size_t oL2 = take(64 * 1024);
  size_t oConv[NLAY], oCond[NLAY], oSkw[NLAY], oOutw[NLAY];
  for (int l = 0; l < NLAY; ++l) {
    oConv[l] = take(192 * 1024);
    oCond[l] = take(48 * 1024);
    oSkw[l] = take(32 * 1024);
    oOutw[l] = take(32 * 1024);
  }
  unsigned short* xA = (unsigned short*)(ws + oXA);
  unsigned short* xB = (unsigned short*)(ws + oXB);
  unsigned short* xin = (unsigned short*)(ws + oXIN);
  unsigned short* c96 = (unsigned short*)(ws + oC);
  float* skip = (float*)(ws + oSK);

  // input conversions (tiled, coalesced both ways through LDS)
  k_prep_x<<<GRIDG, 256, 0, stream>>>(x_in, xin);
  k_prep_c<<<GRIDG, 256, 0, stream>>>(c_in, c96);

  // weight fragment swizzles (args: Kchunks, taps, Kin, srcMstride, srcKstride)
  k_swizzle<<<64, 256, 0, stream>>>(first_w, (unsigned short*)(ws + oFF), 8, 1, 256, 256, 1);
  k_swizzle<<<32, 256, 0, stream>>>(last1_w, (unsigned short*)(ws + oL1), 4, 1, 128, 128, 1);
  k_swizzle<<<64, 256, 0, stream>>>(last2_w, (unsigned short*)(ws + oL2), 4, 1, 128, 128, 1);
  for (int l = 0; l < NLAY; ++l) {
    const int pb = 8 + 8 * l;
    k_swizzle<<<192, 256, 0, stream>>>((const float*)d_in[pb + 0],
                                       (unsigned short*)(ws + oConv[l]), 4, 3, 128, 384, 3);
    k_swizzle<<<48, 256, 0, stream>>>((const float*)d_in[pb + 2],
                                      (unsigned short*)(ws + oCond[l]), 3, 1, 80, 80, 1);
    k_swizzle<<<32, 256, 0, stream>>>((const float*)d_in[pb + 4],
                                      (unsigned short*)(ws + oSkw[l]), 4, 1, 128, 128, 1);
    k_swizzle<<<32, 256, 0, stream>>>((const float*)d_in[pb + 6],
                                      (unsigned short*)(ws + oOutw[l]), 4, 1, 128, 128, 1);
  }

  // first 1x1 conv
  k_first<<<GRIDG, 256, 34816, stream>>>(xin, (const unsigned short*)(ws + oFF),
                                         first_b, xA);

  // 20 fused residual layers (ping-pong activations)
  unsigned short* cur = xA;
  unsigned short* nxt = xB;
  for (int l = 0; l < NLAY; ++l) {
    const int pb = 8 + 8 * l;
    int dil = 1 << (l % (NLAY / 2));
    // skips recurrence folded: s_l contributes r^19 (l==0) else r^(20-l)
    float sc = (float)pow((double)SQH, (l == 0) ? 19 : (20 - l));
    k_layer<<<GRIDG, 256, 155648, stream>>>(
        cur, nxt, c96, skip, (const unsigned short*)(ws + oConv[l]),
        (const unsigned short*)(ws + oCond[l]), (const unsigned short*)(ws + oSkw[l]),
        (const unsigned short*)(ws + oOutw[l]), (const float*)d_in[pb + 1],
        (const float*)d_in[pb + 3], (const float*)d_in[pb + 5],
        (const float*)d_in[pb + 7], dil, sc, l == 0 ? 1 : 0);
    unsigned short* t = cur; cur = nxt; nxt = t;
  }

  // fused head
  k_head<<<GRIDG, 256, 106496, stream>>>(skip, (const unsigned short*)(ws + oL1),
                                         last1_b, (const unsigned short*)(ws + oL2),
                                         last2_b, (float*)d_out);
}
